// NSMMPP_80144089743764
// MI455X (gfx1250) — compile-verified, matrix operand order checked by reading, not run on hardware
//
#include <hip/hip_runtime.h>
#include <hip/hip_bf16.h>
#include <stdint.h>

// ---------------- problem constants ----------------
#define NEV    16384          // N events
#define HD     512            // hidden
#define KL     33             // labels incl. 0
#define KPAD   48             // padded label dim for WMMA N-tiles
#define NSTEP  16383          // N-1 recurrence steps
#define NSIM   16384
#define NBREC  32             // recurrence blocks (one per WGP)
#define ROWS   16             // H rows owned per recurrence block
#define RTHR   128            // threads per recurrence block (4 waves)

// ---------------- small helpers ----------------
typedef __attribute__((ext_vector_type(2)))  __bf16 v2bf;
typedef __attribute__((ext_vector_type(8)))  __bf16 v8bf;
typedef __attribute__((ext_vector_type(16))) __bf16 v16bf;
typedef __attribute__((ext_vector_type(8)))  float  v8f;

__device__ __forceinline__ unsigned short f2bfbits(float f) {
    unsigned u = __builtin_bit_cast(unsigned, f);
    unsigned r = u + 0x7FFFu + ((u >> 16) & 1u);       // RNE
    return (unsigned short)(r >> 16);
}
__device__ __forceinline__ __bf16 f2bf(float f) {
    unsigned short s = f2bfbits(f);
    return __builtin_bit_cast(__bf16, s);
}
__device__ __forceinline__ float sigm(float x) { return 1.f / (1.f + __expf(-x)); }
__device__ __forceinline__ float softplus_s(float x) {
    return fmaxf(x, 0.f) + log1pf(__expf(-fabsf(x)));
}

// ---------------- workspace layout (bytes) ----------------
#define OFF_WBF   0ULL                                   // 7*512*512 bf16 = 3,670,016
#define OFF_UT    3670016ULL                             // 32*256*8*16 u32 = 4,194,304
#define OFF_EMBT  7864320ULL                             // 48*512 bf16    = 49,152
#define OFF_WEMB  7913472ULL                             // 48*7*512 f32   = 688,128
#define OFF_HCUR  8601600ULL                             // 512 f32
#define OFF_BAR   8603648ULL                             // counters
#define OFF_P1    8603904ULL                             // 2048 f32
#define OFF_P2    8612096ULL                             // 2048 f32
#define OFF_ALLH  8620288ULL
#define SZ_ALL    ((unsigned long long)NSTEP * HD * 4ULL) // 33,552,384
#define OFF_ALLC  (OFF_ALLH + SZ_ALL)
#define OFF_ALLCB (OFF_ALLC + SZ_ALL)
#define OFF_ALLD  (OFF_ALLCB + SZ_ALL)
#define OFF_ALLO  (OFF_ALLD + SZ_ALL)

// ============================================================
// kernel 0: zero the cross-kernel mutable state (replay-safe)
// ============================================================
__global__ __launch_bounds__(256) void k_init(float* hcur, unsigned* bar) {
    int t = threadIdx.x;
    for (int i = t; i < HD; i += 256) hcur[i] = 0.f;
    if (t < 8) bar[t] = 0u;
}

// ============================================================
// kernel 1: fp32 -> bf16 conversions + layout transforms
//   Wbf  : [g][h][k] row-major bf16            (WMMA A side)
//   EmbT : [l(48)][k(512)] bf16 (zero padded)  (WMMA B side)
//   Ut   : [blk(32)][k2(256)][g(8)][r(16)] as bf16 pairs
// ============================================================
__global__ __launch_bounds__(256) void k_prep(const float* __restrict__ W,
                                              const float* __restrict__ U,
                                              const float* __restrict__ Emb,
                                              __bf16* __restrict__ Wbf,
                                              uint32_t* __restrict__ Ut,
                                              __bf16* __restrict__ EmbT) {
    const int gsz = gridDim.x * blockDim.x;
    const int gid = blockIdx.x * blockDim.x + threadIdx.x;

    for (int i = gid; i < 7 * HD * HD; i += gsz)
        Wbf[i] = f2bf(W[i]);

    for (int j = gid; j < 7 * HD * (HD / 2); j += gsz) {
        int g   = j / (HD * (HD / 2));
        int rem = j % (HD * (HD / 2));
        int row = rem / (HD / 2);
        int k2  = rem % (HD / 2);
        float f0 = U[((size_t)g * HD + row) * HD + 2 * k2];
        float f1 = U[((size_t)g * HD + row) * HD + 2 * k2 + 1];
        uint32_t wd = (uint32_t)f2bfbits(f0) | ((uint32_t)f2bfbits(f1) << 16);
        int bb = row >> 4, r = row & 15;
        Ut[(((size_t)bb * 256 + k2) * 8 + g) * 16 + r] = wd;
    }

    for (int j = gid; j < KPAD * HD; j += gsz) {
        int l = j / HD, k = j % HD;
        EmbT[j] = (l < KL) ? f2bf(Emb[(size_t)k * KL + l]) : f2bf(0.f);
    }
}

// ============================================================
// kernel 2: WEmb[l][g][row] = sum_k W[g][row][k] * Emb[k][l]
//   bf16 WMMA 16x16x32, fp32 accumulate. 672 tiles = 672 waves.
// ============================================================
__device__ __forceinline__ v16bf ldfrag(const __bf16* p) {
    v8bf a0 = *(const v8bf*)(p);
    v8bf a1 = *(const v8bf*)(p + 16);
    v16bf r;
#pragma unroll
    for (int i = 0; i < 8; ++i) { r[i] = a0[i]; r[i + 8] = a1[i]; }
    return r;
}

__global__ __launch_bounds__(256) void k_wemb(const __bf16* __restrict__ Wbf,
                                              const __bf16* __restrict__ EmbT,
                                              float* __restrict__ WEmb) {
    const int wave = (blockIdx.x * blockDim.x + threadIdx.x) >> 5;  // 0..671
    const int lane = threadIdx.x & 31;
    const int nt = wave % 3;
    const int gm = wave / 3;
    const int mt = gm & 31;
    const int g  = gm >> 5;           // 0..6
    const int m0 = mt * 16, n0 = nt * 16;

    const int idx16 = lane & 15;
    const int ksel  = (lane >> 4) & 1;
    const __bf16* Ab = Wbf  + ((size_t)g * HD + m0 + idx16) * HD + ksel * 8;
    const __bf16* Bb = EmbT + (size_t)(n0 + idx16) * HD + ksel * 8;

    v8f c = {};
#pragma unroll 4
    for (int k0 = 0; k0 < HD; k0 += 32) {
        v16bf a = ldfrag(Ab + k0);
        v16bf b = ldfrag(Bb + k0);
        c = __builtin_amdgcn_wmma_f32_16x16x32_bf16(false, a, false, b,
                                                    (short)0, c, false, false);
    }
    const int n = n0 + idx16;
    const int mbase = m0 + (ksel << 3);
#pragma unroll
    for (int v = 0; v < 8; ++v)
        WEmb[((size_t)n * 7 + g) * HD + (mbase + v)] = c[v];
}

// ============================================================
// kernel 3: the sequential CT-LSTM recurrence.
//   32 blocks x 128 thr; block b owns H rows [16b,16b+16) for all
//   7 gates. U slice (bf16, 128 KB) staged into LDS by async DMA.
//   Per step: 112 dots of length 512, gate math on lanes 0..15,
//   h exchanged via L2 (agent-scope atomics), grid sync via
//   cluster barrier + monotonic-flag atomic barrier.
// ============================================================
__global__ __launch_bounds__(RTHR) void k_recur(
    const uint32_t* __restrict__ Ut, const float* __restrict__ WEmb,
    const float* __restrict__ dvec, const int* __restrict__ label_seq,
    const float* __restrict__ time_seq, float* hcur, unsigned* bar,
    float* __restrict__ all_h, float* __restrict__ all_c,
    float* __restrict__ all_cbar, float* __restrict__ all_delta,
    float* __restrict__ all_o) {
    __shared__ uint32_t Ulds[256 * 8 * 16];   // 131072 B
    __shared__ float    hlds[HD];             // 2048 B
    __shared__ float    templds[7 * ROWS];    // 448 B

    const int tid = threadIdx.x;
    const int b   = blockIdx.x;

    // ---- stage U slice into LDS with async-to-LDS DMA ----
    {
        const uint32_t* src = Ut + (size_t)b * (256 * 8 * 16);
        uint32_t lds_base = (uint32_t)(uintptr_t)(&Ulds[0]);
        for (int i = tid; i < (256 * 8 * 16) / 4; i += RTHR) {
            uint32_t loff  = lds_base + (uint32_t)i * 16u;
            uint64_t gaddr = (uint64_t)(uintptr_t)(src + (size_t)i * 4);
            asm volatile("global_load_async_to_lds_b128 %0, %1, off"
                         :: "v"(loff), "v"(gaddr) : "memory");
        }
#if __has_builtin(__builtin_amdgcn_s_wait_asynccnt)
        __builtin_amdgcn_s_wait_asynccnt(0);
#else
        asm volatile("s_wait_asynccnt 0" ::: "memory");
#endif
    }
    for (int i = tid; i < HD; i += RTHR) hlds[i] = 0.f;
    __syncthreads();

    const int g = tid >> 4;         // 0..7 (gate; 7 == idle)
    const int r = tid & 15;         // row within slice
    const bool active = (g < 7);
    const int row_g = b * ROWS + r;
    const float dreg = active ? dvec[g * HD + row_g] : 0.f;
    float c_s = 0.f, cb_s = 0.f;    // carried state (lanes tid<16)

    const v2bf* Ubf = (const v2bf*)Ulds;      // bf16 pair view of U slice

    for (int t = 0; t < NSTEP; ++t) {
        if (active) {
            float acc0 = 0.f, acc1 = 0.f;
            const v2bf* up = Ubf + (g * 16 + r);
#pragma unroll 8
            for (int k2 = 0; k2 < 256; ++k2) {
                v2bf u = up[(size_t)k2 * 128];
                float2 hv = *(const float2*)&hlds[k2 * 2];
                // fma(fpext(bf16), f32, f32) -> V_FMA_MIX_F32_BF16 pattern
                acc0 = fmaf((float)u.x, hv.x, acc0);
                acc1 = fmaf((float)u.y, hv.y, acc1);
            }
            int lab = label_seq[t];
            templds[g * 16 + r] =
                acc0 + acc1 + dreg + WEmb[((size_t)lab * 7 + g) * HD + row_g];
        }
        __syncthreads();

        if (tid < ROWS) {
            float t0 = templds[0 * 16 + tid];
            float t1 = templds[1 * 16 + tid];
            float t2 = templds[2 * 16 + tid];
            float t3 = templds[3 * 16 + tid];
            float t4 = templds[4 * 16 + tid];
            float t5 = templds[5 * 16 + tid];
            float t6 = templds[6 * 16 + tid];
            float dt = time_seq[t + 1] - time_seq[t];
            float i_g = sigm(t0), f_g = sigm(t1), z = tanhf(t2), o = sigm(t3);
            float i_b = sigm(t4), f_b = sigm(t5), delta = softplus_s(t6);
            float c   = f_g * c_s + i_g * z;
            float cbn = f_b * cb_s + i_b * z;
            float ctn = cbn + (c - cbn) * __expf(-delta * dt);
            float h   = o * tanhf(ctn);
            c_s = ctn; cb_s = cbn;
            size_t o512 = (size_t)t * HD + (size_t)(b * ROWS + tid);
            all_h[o512] = h; all_c[o512] = c; all_cbar[o512] = cbn;
            all_delta[o512] = delta; all_o[o512] = o;
            __hip_atomic_store(&hcur[b * ROWS + tid], h,
                               __ATOMIC_RELAXED, __HIP_MEMORY_SCOPE_AGENT);
        }
        __syncthreads();

        // cluster barrier (NOP when not cluster-dispatched) ...
        __builtin_amdgcn_s_cluster_barrier();
        // ... backed by a monotonic-flag global barrier
        if (tid == 0) {
            __threadfence();
            unsigned old = __hip_atomic_fetch_add(&bar[0], 1u,
                              __ATOMIC_ACQ_REL, __HIP_MEMORY_SCOPE_AGENT);
            if (old == NBREC - 1) {
                __hip_atomic_store(&bar[0], 0u, __ATOMIC_RELAXED,
                                   __HIP_MEMORY_SCOPE_AGENT);
                __hip_atomic_store(&bar[1], (unsigned)(t + 1),
                                   __ATOMIC_RELEASE, __HIP_MEMORY_SCOPE_AGENT);
            } else {
                while (__hip_atomic_load(&bar[1], __ATOMIC_ACQUIRE,
                                         __HIP_MEMORY_SCOPE_AGENT) <
                       (unsigned)(t + 1)) {
                    __builtin_amdgcn_s_sleep(2);
                }
            }
        }
        __syncthreads();

        // reload full h (agent scope => bypass WGP caches)
        {
            float4 hv;
            hv.x = __hip_atomic_load(&hcur[tid * 4 + 0], __ATOMIC_RELAXED, __HIP_MEMORY_SCOPE_AGENT);
            hv.y = __hip_atomic_load(&hcur[tid * 4 + 1], __ATOMIC_RELAXED, __HIP_MEMORY_SCOPE_AGENT);
            hv.z = __hip_atomic_load(&hcur[tid * 4 + 2], __ATOMIC_RELAXED, __HIP_MEMORY_SCOPE_AGENT);
            hv.w = __hip_atomic_load(&hcur[tid * 4 + 3], __ATOMIC_RELAXED, __HIP_MEMORY_SCOPE_AGENT);
            *(float4*)&hlds[tid * 4] = hv;
        }
        __syncthreads();
    }
}

// ============================================================
// kernel 4: term1 partials  (one wave per event row)
// ============================================================
__global__ __launch_bounds__(256) void k_term1(const float* __restrict__ all_h,
                                               const float* __restrict__ w,
                                               const float* __restrict__ log_s,
                                               const int* __restrict__ label_seq,
                                               const int* __restrict__ targetp,
                                               float* __restrict__ partial1) {
    __shared__ float wl[HD];
    __shared__ float acc_s[8];
    const int tid = threadIdx.x;
    const int tgt = *targetp;
    const float* wt = w + (size_t)(tgt - 1) * HD;
    for (int i = tid; i < HD; i += 256) wl[i] = wt[i];
    __syncthreads();
    const float s = __expf(log_s[tgt - 1]);
    const int wv = tid >> 5, lane = tid & 31;
    const int row = blockIdx.x * 8 + wv;
    float contrib = 0.f;
    if (row < NEV - 2) {
        const float* hp = all_h + (size_t)row * HD;
        float a = 0.f;
        for (int k = lane; k < HD; k += 32) a = fmaf(hp[k], wl[k], a);
        for (int off = 16; off; off >>= 1) a += __shfl_xor(a, off, 32);
        if (lane == 0 && label_seq[row + 1] == tgt) {
            float lam = s * softplus_s(a / s) + 1e-9f;
            contrib = logf(lam);
        }
    }
    if (lane == 0) acc_s[wv] = contrib;
    __syncthreads();
    if (tid == 0) {
        float tsum = 0.f;
        for (int i = 0; i < 8; ++i) tsum += acc_s[i];
        partial1[blockIdx.x] = tsum;
    }
}

// ============================================================
// kernel 5: term2 partials  (one wave per MC sample)
// ============================================================
__global__ __launch_bounds__(256) void k_term2(
    const float* __restrict__ all_c, const float* __restrict__ all_cbar,
    const float* __restrict__ all_delta, const float* __restrict__ all_o,
    const float* __restrict__ sim_time_seq, const int* __restrict__ sim_time_idx,
    const float* __restrict__ time_seq, const float* __restrict__ w,
    const float* __restrict__ log_s, const int* __restrict__ targetp,
    float* __restrict__ partial2) {
    __shared__ float wl[HD];
    __shared__ float acc_s[8];
    const int tid = threadIdx.x;
    const int tgt = *targetp;
    const float* wt = w + (size_t)(tgt - 1) * HD;
    for (int i = tid; i < HD; i += 256) wl[i] = wt[i];
    __syncthreads();
    const float s = __expf(log_s[tgt - 1]);
    const int wv = tid >> 5, lane = tid & 31;
    const int j = blockIdx.x * 8 + wv;           // 0..16383 exactly
    const int idx = sim_time_idx[j];
    const float td = sim_time_seq[j] - time_seq[idx];
    const size_t base = (size_t)idx * HD;
    float a = 0.f;
    for (int k = lane; k < HD; k += 32) {
        float c  = all_c[base + k];
        float cb = all_cbar[base + k];
        float de = all_delta[base + k];
        float oo = all_o[base + k];
        float cs = cb + (c - cb) * __expf(-de * td);
        a = fmaf(oo * tanhf(cs), wl[k], a);
    }
    for (int off = 16; off; off >>= 1) a += __shfl_xor(a, off, 32);
    if (lane == 0) {
        float lam = s * softplus_s(a / s) + 1e-9f;
        acc_s[wv] = lam;
    }
    __syncthreads();
    if (tid == 0) {
        float tsum = 0.f;
        for (int i = 0; i < 8; ++i) tsum += acc_s[i];
        partial2[blockIdx.x] = tsum;
    }
}

// ============================================================
// kernel 6: deterministic final reduction
// ============================================================
__global__ void k_final(const float* __restrict__ partial1,
                        const float* __restrict__ partial2,
                        const float* __restrict__ time_seq,
                        float* __restrict__ out) {
    if (threadIdx.x == 0 && blockIdx.x == 0) {
        float t1 = 0.f, t2 = 0.f;
        for (int i = 0; i < 2048; ++i) t1 += partial1[i];
        for (int i = 0; i < 2048; ++i) t2 += partial2[i];
        float mean = t2 / (float)NSIM;
        float comp = mean * (time_seq[NEV - 1] - time_seq[0]);
        out[0] = -(t1 - comp);
    }
}

// ============================================================
extern "C" void kernel_launch(void* const* d_in, const int* in_sizes, int n_in,
                              void* d_out, int out_size, void* d_ws, size_t ws_size,
                              hipStream_t stream) {
    const int*   label_seq    = (const int*)d_in[0];
    const float* time_seq     = (const float*)d_in[1];
    const float* sim_time_seq = (const float*)d_in[2];
    const int*   sim_time_idx = (const int*)d_in[3];
    const float* Emb          = (const float*)d_in[4];
    const float* W            = (const float*)d_in[5];
    const float* U            = (const float*)d_in[6];
    const float* dvec         = (const float*)d_in[7];
    const float* w            = (const float*)d_in[8];
    const float* log_s        = (const float*)d_in[9];
    const int*   target       = (const int*)d_in[10];

    char* ws = (char*)d_ws;
    __bf16*   Wbf   = (__bf16*)(ws + OFF_WBF);
    uint32_t* Ut    = (uint32_t*)(ws + OFF_UT);
    __bf16*   EmbT  = (__bf16*)(ws + OFF_EMBT);
    float*    WEmb  = (float*)(ws + OFF_WEMB);
    float*    hcur  = (float*)(ws + OFF_HCUR);
    unsigned* bar   = (unsigned*)(ws + OFF_BAR);
    float*    p1    = (float*)(ws + OFF_P1);
    float*    p2    = (float*)(ws + OFF_P2);
    float*    all_h = (float*)(ws + OFF_ALLH);
    float*    all_c = (float*)(ws + OFF_ALLC);
    float*    all_cb= (float*)(ws + OFF_ALLCB);
    float*    all_d = (float*)(ws + OFF_ALLD);
    float*    all_o = (float*)(ws + OFF_ALLO);

    k_init<<<1, 256, 0, stream>>>(hcur, bar);
    k_prep<<<1792, 256, 0, stream>>>(W, U, Emb, Wbf, Ut, EmbT);
    k_wemb<<<84, 256, 0, stream>>>(Wbf, EmbT, WEmb);
    k_recur<<<NBREC, RTHR, 0, stream>>>(Ut, WEmb, dvec, label_seq, time_seq,
                                        hcur, bar, all_h, all_c, all_cb,
                                        all_d, all_o);
    k_term1<<<2048, 256, 0, stream>>>(all_h, w, log_s, label_seq, target, p1);
    k_term2<<<2048, 256, 0, stream>>>(all_c, all_cb, all_d, all_o,
                                      sim_time_seq, sim_time_idx, time_seq,
                                      w, log_s, target, p2);
    k_final<<<1, 32, 0, stream>>>(p1, p2, time_seq, (float*)d_out);
}
